// MultiHeadAttention_89524298318221
// MI455X (gfx1250) — matrix operand order repaired and checked by reading, not compiled, not run on hardware
//
#include <hip/hip_runtime.h>

// ---------------------------------------------------------------------------
// MHA for MI455X (gfx1250): bf16 WMMA everywhere, flash-attention inner loop
// with TDM (tensor_load_to_lds) double-buffered K/V staging.
// B=2, S=SQ=2048, D_MODEL=1024, H=16, DEPTH=64.
// ---------------------------------------------------------------------------

typedef __bf16 v16bf __attribute__((ext_vector_type(16)));
typedef __bf16 v8bf  __attribute__((ext_vector_type(8)));
typedef float  v8f   __attribute__((ext_vector_type(8)));
typedef unsigned int u32x4 __attribute__((ext_vector_type(4)));
typedef int          i32x8 __attribute__((ext_vector_type(8)));
typedef int          i32x4 __attribute__((ext_vector_type(4)));

constexpr int Bc  = 2;
constexpr int Sc  = 2048;
constexpr int SQc = 2048;
constexpr int Hc  = 16;
constexpr int Dm  = 1024;   // d_model
constexpr int Dh  = 64;     // depth per head

// WMMA fragment layouts used here (wave32, 16x16x32 bf16; A/B k order is this file's choice):
//  A (16xK=32): lane row = lane&15; element e -> K = 16*(lane>>4) + e
//               => one contiguous 16-elem run (read as two 8-elem loads).
//  B (K=32x16): lane col = lane&15; element e -> K = 16*(lane>>4) + e
//               => one contiguous 16-elem run.
//  C/D (16x16 f32): lane col = lane&15; element i -> row = i + 8*(lane>>4).

// ---------------------------------------------------------------------------
// TDM: issue a 2D tile load (bf16 elements) global -> LDS via the Tensor
// Data Mover. D# built per CDNA5 ISA §8 (group0/group1, 2D, no padding,
// no gather, no multicast). Tracked by TENSORcnt.
// ---------------------------------------------------------------------------
__device__ __forceinline__ void tdm_load_2d_bf16(const void* gptr, void* ldsptr,
                                                 unsigned tensor_d0, unsigned tensor_d1,
                                                 unsigned tile_d0,   unsigned tile_d1,
                                                 unsigned stride0)
{
    unsigned long long ga = (unsigned long long)gptr;
    unsigned lds = (unsigned)(unsigned long long)ldsptr;   // LDS byte offset (low 32b)

    u32x4 g0;
    g0[0] = 1u;                                            // count=1, user D#
    g0[1] = lds;                                           // lds_addr
    g0[2] = (unsigned)(ga & 0xffffffffu);                  // global_addr[31:0]
    g0[3] = (unsigned)((ga >> 32) & 0x01ffffffu) | (2u << 30);  // addr[56:32] | type=2

    i32x8 g1;
    g1[0] = (int)(1u << 16);                               // data_size=1 -> 2 bytes
    g1[1] = (int)((tensor_d0 & 0xffffu) << 16);            // tensor_dim0[15:0]
    g1[2] = (int)(((tensor_d0 >> 16) & 0xffffu) |
                  ((tensor_d1 & 0xffffu) << 16));          // tensor_dim0[31:16]|dim1[15:0]
    g1[3] = (int)(((tensor_d1 >> 16) & 0xffffu) |
                  ((tile_d0 & 0xffffu) << 16));            // tensor_dim1[31:16]|tile_dim0
    g1[4] = (int)(tile_d1 & 0xffffu);                      // tile_dim1 (tile_dim2=0)
    g1[5] = (int)stride0;                                  // tensor_dim0_stride[31:0]
    g1[6] = 0;                                             // stride0[47:32]|stride1[15:0]
    g1[7] = 0;                                             // stride1[47:16]

    i32x4 z4 = {0, 0, 0, 0};
#if defined(__clang_major__) && (__clang_major__ >= 23)
    i32x8 z8 = {0, 0, 0, 0, 0, 0, 0, 0};
    __builtin_amdgcn_tensor_load_to_lds(g0, g1, z4, z4, z8, 0);
#else
    __builtin_amdgcn_tensor_load_to_lds(g0, g1, z4, z4, 0);
#endif
}

// ---------------------------------------------------------------------------
// Tiled GEMM: Y = X @ W + bias.   X:[M,1024] (f32 or bf16), W:[1024,1024] f32.
// OUT_MODE 0: bf16, head-split [B,H,S,64]
// OUT_MODE 1: bf16, head-split transposed [B,H,64,S]   (for V^T)
// OUT_MODE 2: f32, flat [M,1024]                        (final output)
// Block: 256 threads (8 waves). Block tile 128(M) x 64(N). K-step 32.
// ---------------------------------------------------------------------------
template<int IN_BF16, int OUT_MODE>
__global__ __launch_bounds__(256)
void proj_gemm(const void* __restrict__ Xv,
               const float* __restrict__ W,
               const float* __restrict__ bias,
               void* __restrict__ Yv,
               int S_tok)
{
    const int lane = threadIdx.x & 31;
    const int wave = threadIdx.x >> 5;
    const int hi   = lane >> 4;
    const int lo16 = lane & 15;
    const int n0   = blockIdx.y * 64;
    const int r0   = blockIdx.x * 128 + wave * 16;

    // W tile staged transposed as bf16: Wt[n][k], row stride 48 (pad, 32B-mult)
    __shared__ __align__(32) __bf16 Wt[64 * 48];

    v8f acc[4] = {};

    for (int kk = 0; kk < Dm; kk += 32) {
        __syncthreads();
        // stage 32(K) x 64(N) fp32 -> bf16 transposed into LDS
        #pragma unroll
        for (int j = 0; j < 8; ++j) {
            int idx = threadIdx.x + 256 * j;     // 2048 elements
            int k = idx >> 6;                    // 0..31
            int n = idx & 63;                    // 0..63
            float w = W[(size_t)(kk + k) * Dm + (n0 + n)];
            Wt[n * 48 + k] = (__bf16)w;
        }
        __syncthreads();

        // A fragment (16 x 32) from X, converted to bf16 in registers
        v16bf a;
        {
            const int row = r0 + lo16;
            if (IN_BF16) {
                const __bf16* X = (const __bf16*)Xv;
                const __bf16* Xr = X + (size_t)row * Dm + kk;
                v8bf x0 = *(const v8bf*)(Xr + 16 * hi);
                v8bf x1 = *(const v8bf*)(Xr + 16 * hi + 8);
                #pragma unroll
                for (int e = 0; e < 8; ++e) { a[e] = x0[e]; a[8 + e] = x1[e]; }
            } else {
                const float* X = (const float*)Xv;
                const float* Xr = X + (size_t)row * Dm + kk;
                float4 f0 = ((const float4*)(Xr + 16 * hi))[0];
                float4 f1 = ((const float4*)(Xr + 16 * hi))[1];
                float4 f2 = ((const float4*)(Xr + 16 * hi + 8))[0];
                float4 f3 = ((const float4*)(Xr + 16 * hi + 8))[1];
                float t[16] = { f0.x, f0.y, f0.z, f0.w, f1.x, f1.y, f1.z, f1.w,
                                f2.x, f2.y, f2.z, f2.w, f3.x, f3.y, f3.z, f3.w };
                #pragma unroll
                for (int e = 0; e < 16; ++e) a[e] = (__bf16)t[e];
            }
        }

        // 4 x (16x16) output tiles across N
        #pragma unroll
        for (int t = 0; t < 4; ++t) {
            const v16bf b = *(const v16bf*)&Wt[(16 * t + lo16) * 48 + 16 * hi];
            acc[t] = __builtin_amdgcn_wmma_f32_16x16x32_bf16(
                false, a, false, b, (short)0, acc[t], false, false);
        }
    }

    // epilogue: bias + store
    #pragma unroll
    for (int t = 0; t < 4; ++t) {
        const int n = n0 + 16 * t + lo16;
        const float bv = bias[n];
        #pragma unroll
        for (int i = 0; i < 8; ++i) {
            const int row = r0 + i + 8 * hi;          // global token row
            float v = acc[t][i] + bv;
            if (OUT_MODE == 2) {
                ((float*)Yv)[(size_t)row * Dm + n] = v;
            } else {
                const int b = row / S_tok, s = row % S_tok;
                const int h = n >> 6, d = n & 63;
                __bf16* Y = (__bf16*)Yv;
                if (OUT_MODE == 0)
                    Y[((size_t)(b * Hc + h) * S_tok + s) * Dh + d] = (__bf16)v;
                else
                    Y[((size_t)(b * Hc + h) * Dh + d) * S_tok + s] = (__bf16)v;
            }
        }
    }
}

// ---------------------------------------------------------------------------
// Flash attention. Q:[B,H,SQ,64] bf16, K:[B,H,S,64] bf16, Vt:[B,H,64,S] bf16.
// O (flat bf16 [B*SQ, 1024]). One wave = 16 query rows; 32 keys per iter.
// K/V tiles staged into double-buffered LDS by the TDM (wave 0 issues,
// TENSORcnt-waited, barrier-published); fragments fed from ds_load_b128.
// ---------------------------------------------------------------------------
__global__ __launch_bounds__(256)
void flash_attn(const void* __restrict__ Qv, const void* __restrict__ Kv,
                const void* __restrict__ Vtv, const void* __restrict__ maskv,
                void* __restrict__ Ov)
{
    const __bf16* Q  = (const __bf16*)Qv;
    const __bf16* K  = (const __bf16*)Kv;
    const __bf16* Vt = (const __bf16*)Vtv;
    const unsigned char* mask = (const unsigned char*)maskv;
    __bf16* O = (__bf16*)Ov;

    const int lane = threadIdx.x & 31;
    const int wave = threadIdx.x >> 5;
    const int hi   = lane >> 4;
    const int lo16 = lane & 15;
    const int bh   = blockIdx.y;          // b*H + h
    const int b    = bh / Hc;
    const int h    = bh % Hc;
    const int q0   = blockIdx.x * 128 + wave * 16;

    // double-buffered K (32x64) and V^T (64x32) tiles + per-wave P tiles
    __shared__ __align__(128) __bf16 Ktile[2][32 * 64];   // 2 x 4KB
    __shared__ __align__(128) __bf16 Vtile[2][64 * 32];   // 2 x 4KB
    __shared__ __align__(32)  __bf16 Plds[8][16 * 32];    // 8KB

    const __bf16* Kbase  = K  + (size_t)bh * Sc * Dh;
    const __bf16* Vtbase = Vt + (size_t)bh * Dh * Sc;

    // Q A-fragments (depth 0..31 and 32..63), loaded once
    v16bf qa[2];
    {
        const __bf16* Qr = Q + ((size_t)bh * SQc + q0 + lo16) * Dh;
        #pragma unroll
        for (int kk = 0; kk < 2; ++kk) {
            v8bf x0 = *(const v8bf*)(Qr + 32 * kk + 16 * hi);
            v8bf x1 = *(const v8bf*)(Qr + 32 * kk + 16 * hi + 8);
            #pragma unroll
            for (int e = 0; e < 8; ++e) { qa[kk][e] = x0[e]; qa[kk][8 + e] = x1[e]; }
        }
    }

    v8f o[4] = {};
    float mrow[8], lrow[8];
    #pragma unroll
    for (int i = 0; i < 8; ++i) { mrow[i] = -INFINITY; lrow[i] = 0.0f; }

    // prologue: TDM-stage tile 0 into buffer 0
    int buf = 0;
    if (wave == 0) {
        tdm_load_2d_bf16(Kbase, &Ktile[0][0],
                         /*tensor_d0=*/Dh, /*tensor_d1=*/Sc,
                         /*tile_d0=*/Dh,   /*tile_d1=*/32, /*stride0=*/Dh);
        tdm_load_2d_bf16(Vtbase, &Vtile[0][0],
                         /*tensor_d0=*/Sc, /*tensor_d1=*/Dh,
                         /*tile_d0=*/32,   /*tile_d1=*/Dh, /*stride0=*/Sc);
        __builtin_amdgcn_s_wait_tensorcnt(0);
    }
    __syncthreads();

    for (int s = 0; s < Sc; s += 32) {
        const int nbuf = buf ^ 1;
        // overlap: kick off DMA of the next K/V tile while computing this one
        if (wave == 0 && (s + 32) < Sc) {
            tdm_load_2d_bf16(Kbase + (size_t)(s + 32) * Dh, &Ktile[nbuf][0],
                             Dh, Sc, Dh, 32, Dh);
            tdm_load_2d_bf16(Vtbase + (s + 32), &Vtile[nbuf][0],
                             Sc, Dh, 32, Dh, Sc);
        }

        // ---- S = (Q @ K^T) * 1/sqrt(64), masked. Two 16x16 C tiles. ----
        v8f sc[2] = {};
        #pragma unroll
        for (int t = 0; t < 2; ++t) {
            #pragma unroll
            for (int kk = 0; kk < 2; ++kk) {
                const v16bf kb = *(const v16bf*)
                    &Ktile[buf][(16 * t + lo16) * Dh + 32 * kk + 16 * hi];
                sc[t] = __builtin_amdgcn_wmma_f32_16x16x32_bf16(
                    false, qa[kk], false, kb, (short)0, sc[t], false, false);
            }
            const bool keep = mask[b * Sc + s + 16 * t + lo16] != 0;
            #pragma unroll
            for (int i = 0; i < 8; ++i) {
                float v = sc[t][i] * 0.125f;
                sc[t][i] = keep ? v : -1e30f;
            }
        }

        // ---- online softmax (row stats live per-lane: rows i + 8*hi) ----
        float pr[2][8], alpha[8];
        #pragma unroll
        for (int i = 0; i < 8; ++i) {
            float v = fmaxf(sc[0][i], sc[1][i]);
            #pragma unroll
            for (int m = 1; m < 16; m <<= 1) v = fmaxf(v, __shfl_xor(v, m, 32));
            const float mnew = fmaxf(mrow[i], v);
            alpha[i] = __expf(mrow[i] - mnew);
            pr[0][i] = __expf(sc[0][i] - mnew);
            pr[1][i] = __expf(sc[1][i] - mnew);
            float rs = pr[0][i] + pr[1][i];
            #pragma unroll
            for (int m = 1; m < 16; m <<= 1) rs += __shfl_xor(rs, m, 32);
            lrow[i] = lrow[i] * alpha[i] + rs;
            mrow[i] = mnew;
        }

        // rescale running output
        #pragma unroll
        for (int dt = 0; dt < 4; ++dt)
            #pragma unroll
            for (int i = 0; i < 8; ++i) o[dt][i] *= alpha[i];

        // ---- P (C-layout) -> LDS -> A-fragment ----
        #pragma unroll
        for (int t = 0; t < 2; ++t)
            #pragma unroll
            for (int i = 0; i < 8; ++i)
                Plds[wave][(i + 8 * hi) * 32 + 16 * t + lo16] = (__bf16)pr[t][i];

        v16bf pa;
        {
            const __bf16* Pw = &Plds[wave][lo16 * 32];
            v8bf p0 = *(const v8bf*)(Pw + 16 * hi);
            v8bf p1 = *(const v8bf*)(Pw + 16 * hi + 8);
            #pragma unroll
            for (int e = 0; e < 8; ++e) { pa[e] = p0[e]; pa[8 + e] = p1[e]; }
        }

        // ---- O += P @ V  (4 depth tiles of 16) ----
        #pragma unroll
        for (int dt = 0; dt < 4; ++dt) {
            const v16bf vb = *(const v16bf*)
                &Vtile[buf][(16 * dt + lo16) * 32 + 16 * hi];
            o[dt] = __builtin_amdgcn_wmma_f32_16x16x32_bf16(
                false, pa, false, vb, (short)0, o[dt], false, false);
        }

        // rotate buffers: wait for the in-flight DMA, publish to all waves
        if (wave == 0) __builtin_amdgcn_s_wait_tensorcnt(0);
        __syncthreads();
        buf = nbuf;
    }

    // ---- normalize and store bf16 O (flat [B*SQ, 1024], head-concat) ----
    float inv[8];
    #pragma unroll
    for (int i = 0; i < 8; ++i) inv[i] = (lrow[i] > 0.0f) ? 1.0f / lrow[i] : 0.0f;
    #pragma unroll
    for (int dt = 0; dt < 4; ++dt) {
        const int feat = h * Dh + 16 * dt + lo16;
        #pragma unroll
        for (int i = 0; i < 8; ++i) {
            const size_t row = (size_t)b * SQc + q0 + i + 8 * hi;
            O[row * Dm + feat] = (__bf16)(o[dt][i] * inv[i]);
        }
    }
}

// ---------------------------------------------------------------------------
// Host launcher.
// Inputs: 0 value, 1 key, 2 query, 3 key_padding_mask,
//         4 Wq, 5 bq, 6 Wk, 7 bk, 8 Wv, 9 bv, 10 Wo, 11 bo.
// Workspace: Q bf16 8MB | K bf16 8MB | V^T bf16 8MB | O bf16 8MB.
// ---------------------------------------------------------------------------
extern "C" void kernel_launch(void* const* d_in, const int* in_sizes, int n_in,
                              void* d_out, int out_size, void* d_ws, size_t ws_size,
                              hipStream_t stream)
{
    const float* value = (const float*)d_in[0];
    const float* keyi  = (const float*)d_in[1];
    const float* query = (const float*)d_in[2];
    const void*  mask  = d_in[3];
    const float* Wq = (const float*)d_in[4];  const float* bq = (const float*)d_in[5];
    const float* Wk = (const float*)d_in[6];  const float* bk = (const float*)d_in[7];
    const float* Wv = (const float*)d_in[8];  const float* bv = (const float*)d_in[9];
    const float* Wo = (const float*)d_in[10]; const float* bo = (const float*)d_in[11];

    char* ws = (char*)d_ws;
    void* Qh  = (void*)(ws);
    void* Kh  = (void*)(ws + ((size_t)8 << 20));
    void* Vth = (void*)(ws + ((size_t)16 << 20));
    void* Oh  = (void*)(ws + ((size_t)24 << 20));

    dim3 blk(256);
    dim3 gp((Bc * Sc) / 128, Dm / 64);          // 32 x 16 workgroups

    proj_gemm<0, 0><<<gp, blk, 0, stream>>>(query, Wq, bq, Qh, SQc);
    proj_gemm<0, 0><<<gp, blk, 0, stream>>>(keyi,  Wk, bk, Kh, Sc);
    proj_gemm<0, 1><<<gp, blk, 0, stream>>>(value, Wv, bv, Vth, Sc);

    dim3 gf(SQc / 128, Bc * Hc);                // 16 x 32 workgroups
    flash_attn<<<gf, blk, 0, stream>>>(Qh, Kh, Vth, mask, Oh);

    proj_gemm<1, 2><<<gp, blk, 0, stream>>>(Oh, Wo, bo, d_out, SQc);
}
